// CrossAttention_5961414607456
// MI455X (gfx1250) — compile-verified, matrix operand order checked by reading, not run on hardware
//
#include <hip/hip_runtime.h>
#include <hip/hip_bf16.h>

// ---------------------------------------------------------------------------
// CrossAttention (Restormer-style transposed channel attention) for MI455X.
// Memory-bound pipeline; WMMA f16 (v_wmma_f32_16x16x32_f16) for all GEMMs.
// Round 2: double-buffered LDS staging in the GEMM; attention-score GEMM
// split over 16 K-chunks (64 -> 1024 blocks) + tiny finalize/softmax kernel.
// ---------------------------------------------------------------------------

typedef _Float16 h8  __attribute__((ext_vector_type(8)));
typedef _Float16 h16 __attribute__((ext_vector_type(16)));
typedef float    f8  __attribute__((ext_vector_type(8)));
typedef float    f4  __attribute__((ext_vector_type(4)));

#define HW   16384
#define C    192
#define D2   384
#define HEADS 8
#define HCH  24   // channels per head
#define BP   40   // LDS B-tile pitch in halves (80B, keeps b128 aligned)

__device__ __forceinline__ h16 cat8(h8 lo, h8 hi) {
    return __builtin_shufflevector(lo, hi, 0,1,2,3,4,5,6,7,8,9,10,11,12,13,14,15);
}

// ---------------------------------------------------------------------------
// Y[b,M,n] = W[M,K] @ X[b,K,n]   (1x1 conv as GEMM, N = HW)
// Block: 256 thr (8 waves). Block tile 32(M) x 256(N). Wave tile 16 x 64.
// Weights (32 x K, f16) parked in LDS for the whole kernel. X tiles staged
// transposed ([pixel][k], k-contiguous) with DOUBLE BUFFERING: stage tile
// k+32 while issuing WMMAs on tile k -> one barrier per k-step.
// ---------------------------------------------------------------------------
__global__ void gemm1x1(const float* __restrict__ X, const float* __restrict__ W,
                        float* __restrict__ Y, int M, int K) {
    extern __shared__ _Float16 smem[];
    _Float16* ldsA = smem;            // 32 x K (f16)
    _Float16* ldsB = smem + 32 * K;   // 2 x (256 x BP) f16, transposed tiles

    const int t    = threadIdx.x;
    const int wave = t >> 5, lane = t & 31;
    const int b    = blockIdx.z;
    const int n0   = blockIdx.x * 256;
    const int m0   = blockIdx.y * 32;

    // park weights (fp32 -> f16) in LDS once
    for (int i = t; i < 32 * K; i += 256)
        ldsA[i] = (_Float16)W[(size_t)(m0 + (i / K)) * K + (i % K)];

    const float* Xb = X + (size_t)b * K * HW;

    const int mw  = (wave & 1) * 16;   // wave M offset inside block tile
    const int nw  = (wave >> 1) * 64;  // wave N offset inside block tile
    const int row = lane & 15;
    const int klo = (lane >> 4) * 8;   // lanes 0-15: K{0..7,16..23}; 16-31: +8

    auto stage = [&](int buf, int k0) {
        _Float16* dst = ldsB + buf * (256 * BP);
        for (int kk = 0; kk < 32; ++kk)
            dst[t * BP + kk] = (_Float16)Xb[(size_t)(k0 + kk) * HW + n0 + t];
    };

    f8 acc[4] = {};

    stage(0, 0);
    __syncthreads();                   // covers ldsA + first B tile

    int buf = 0;
    for (int k0 = 0; k0 < K; k0 += 32, buf ^= 1) {
        if (k0 + 32 < K) stage(buf ^ 1, k0 + 32);   // prefetch next tile

        const h8* pa0 = (const h8*)&ldsA[(size_t)(mw + row) * K + k0 + klo];
        const h8* pa1 = (const h8*)&ldsA[(size_t)(mw + row) * K + k0 + klo + 16];
        h16 afr = cat8(*pa0, *pa1);

        const _Float16* Bcur = ldsB + buf * (256 * BP);
#pragma unroll
        for (int j = 0; j < 4; ++j) {
            int px = nw + j * 16 + row;
            const h8* pb0 = (const h8*)&Bcur[px * BP + klo];
            const h8* pb1 = (const h8*)&Bcur[px * BP + klo + 16];
            h16 bfr = cat8(*pb0, *pb1);
            acc[j] = __builtin_amdgcn_wmma_f32_16x16x32_f16(
                false, afr, false, bfr, (short)0, acc[j], false, false);
        }
        __syncthreads();               // staged tile visible; cur tile free
    }

    // C/D layout: lane<16 -> M=r, lane>=16 -> M=r+8; N = lane&15
    const int rbase = mw + (lane >> 4) * 8;
#pragma unroll
    for (int j = 0; j < 4; ++j) {
        int col = n0 + nw + j * 16 + (lane & 15);
#pragma unroll
        for (int r = 0; r < 8; ++r)
            Y[((size_t)b * M + (m0 + rbase + r)) * HW + col] = acc[j][r];
    }
}

// ---------------------------------------------------------------------------
// 3x3 grouped conv, pad=1.  inPerGroup==2 -> q_dwconv (groups=192),
// inPerGroup==1 -> depthwise (groups=384).  X,Y: [b, 384, 128, 128].
// ---------------------------------------------------------------------------
__global__ void dwconv3x3(const float* __restrict__ X, const float* __restrict__ Wd,
                          float* __restrict__ Y, int inPerGroup) {
    size_t idx = (size_t)blockIdx.x * 256 + threadIdx.x;   // b*384*HW elements
    int n    = (int)(idx & (HW - 1));
    int bc   = (int)(idx >> 14);          // b*384 + ch
    int ch   = bc % D2;
    int b384 = bc - ch;
    int hh = n >> 7, wc = n & 127;
    int inBase = (inPerGroup == 2) ? (ch & ~1) : ch;

    float acc = 0.0f;
    for (int i = 0; i < inPerGroup; ++i) {
        const float* xin = X + ((size_t)(b384 + inBase + i) << 14);
        const float* wk  = Wd + (size_t)(ch * inPerGroup + i) * 9;
#pragma unroll
        for (int kh = 0; kh < 3; ++kh) {
            int y = hh + kh - 1;
            if (y < 0 || y > 127) continue;
#pragma unroll
            for (int kw = 0; kw < 3; ++kw) {
                int x = wc + kw - 1;
                if (x < 0 || x > 127) continue;
                acc += xin[(y << 7) + x] * wk[kh * 3 + kw];
            }
        }
    }
    Y[idx] = acc;
}

// ---------------------------------------------------------------------------
// Inverse L2 norms per row (row = one channel's HW plane).
// norms layout per batch (768 rows): [0,384)=q_ rows, [384,576)=k1, [576,768)=k2
// ---------------------------------------------------------------------------
__global__ void rownorms(const float* __restrict__ qB, const float* __restrict__ kv1,
                         const float* __restrict__ kv2, float* __restrict__ norms) {
    int r = blockIdx.x;
    int b = r / 768, rr = r % 768;
    const float* src;
    if (rr < 384)      src = qB  + ((size_t)(b * D2 + rr) << 14);
    else if (rr < 576) src = kv1 + ((size_t)(b * D2 + (rr - 384)) << 14);
    else               src = kv2 + ((size_t)(b * D2 + (rr - 576)) << 14);

    int t = threadIdx.x;
    float s = 0.0f;
    for (int i = t; i < HW; i += 256) { float v = src[i]; s += v * v; }
    __shared__ float red[256];
    red[t] = s; __syncthreads();
    for (int off = 128; off > 0; off >>= 1) {
        if (t < off) red[t] += red[t + off];
        __syncthreads();
    }
    if (t == 0) norms[r] = 1.0f / fmaxf(sqrtf(red[0]), 1e-12f);
}

// pack a 16-wide f16 fragment from a global fp32 row, scaled (l2norm folded in)
__device__ __forceinline__ h16 load_frag_scaled(const float* rowptr, int n0, int klo, float s) {
    f4 x0 = *(const f4*)(rowptr + n0 + klo);
    f4 x1 = *(const f4*)(rowptr + n0 + klo + 4);
    f4 x2 = *(const f4*)(rowptr + n0 + klo + 16);
    f4 x3 = *(const f4*)(rowptr + n0 + klo + 20);
    h16 r;
#pragma unroll
    for (int j = 0; j < 4; ++j) {
        r[j]      = (_Float16)(x0[j] * s);
        r[4 + j]  = (_Float16)(x1[j] * s);
        r[8 + j]  = (_Float16)(x2[j] * s);
        r[12 + j] = (_Float16)(x3[j] * s);
    }
    return r;
}

// ---------------------------------------------------------------------------
// Attention scores, pass 1: partial q_hat . k_hat^T over a 1024-wide n-chunk.
// Grid (16 chunks, head+8*br, b) = 1024 blocks; 8 waves each own 128 pixels.
// Writes f32 partial 32x32 tiles to ws (rows/cols >= 24 accumulate zeros).
// ---------------------------------------------------------------------------
__global__ void attn_partial(const float* __restrict__ qB, const float* __restrict__ kv1,
                             const float* __restrict__ kv2, const float* __restrict__ norms,
                             float* __restrict__ partials) {
    const int chunk = blockIdx.x;                 // 0..15
    const int y     = blockIdx.y;                 // head + HEADS*br
    const int head  = y & 7, br = y >> 3;
    const int b     = blockIdx.z;
    const int t = threadIdx.x, wave = t >> 5, lane = t & 31;

    const float* qptr = qB + ((size_t)(b * D2 + br * C + head * HCH)) * HW;
    const float* kptr = (br ? kv2 : kv1) + ((size_t)(b * D2 + head * HCH)) * HW;
    const float* qs = norms + b * 768 + br * C + head * HCH;
    const float* ks = norms + b * 768 + 384 + br * C + head * HCH;

    const int row = lane & 15;
    const int klo = (lane >> 4) * 8;

    const float* qrp[2]; float qsc[2];
    const float* krp[2]; float ksc[2];
#pragma unroll
    for (int mt = 0; mt < 2; ++mt) {
        int c  = mt * 16 + row;
        int ca = (c < HCH) ? c : 0;               // clamp addr, zero scale
        qrp[mt] = qptr + (size_t)ca * HW;
        krp[mt] = kptr + (size_t)ca * HW;
        qsc[mt] = (c < HCH) ? qs[ca] : 0.0f;
        ksc[mt] = (c < HCH) ? ks[ca] : 0.0f;
    }

    f8 acc[4] = {};
    const int nbase = chunk * 1024 + wave * 128;
    for (int n0 = nbase; n0 < nbase + 128; n0 += 32) {
        h16 af[2], bf[2];
#pragma unroll
        for (int mt = 0; mt < 2; ++mt) af[mt] = load_frag_scaled(qrp[mt], n0, klo, qsc[mt]);
#pragma unroll
        for (int nt = 0; nt < 2; ++nt) bf[nt] = load_frag_scaled(krp[nt], n0, klo, ksc[nt]);
#pragma unroll
        for (int mt = 0; mt < 2; ++mt)
#pragma unroll
            for (int nt = 0; nt < 2; ++nt)
                acc[mt * 2 + nt] = __builtin_amdgcn_wmma_f32_16x16x32_f16(
                    false, af[mt], false, bf[nt], (short)0, acc[mt * 2 + nt], false, false);
    }

    // cross-wave reduction of the 32x32 partial sums
    __shared__ float red[8][1024];
#pragma unroll
    for (int mt = 0; mt < 2; ++mt)
#pragma unroll
        for (int nt = 0; nt < 2; ++nt)
#pragma unroll
            for (int r = 0; r < 8; ++r) {
                int rr = mt * 16 + (lane >> 4) * 8 + r;
                int cc = nt * 16 + (lane & 15);
                red[wave][rr * 32 + cc] = acc[mt * 2 + nt][r];
            }
    __syncthreads();
    float* dst = partials + (((size_t)(b * 16 + y)) * 16 + chunk) * 1024;
    for (int i = t; i < 1024; i += 256) {
        float s = 0;
#pragma unroll
        for (int w = 0; w < 8; ++w) s += red[w][i];
        dst[i] = s;
    }
}

// ---------------------------------------------------------------------------
// Attention scores, pass 2: reduce 16 chunk-partials, *temp, softmax over 24,
// store padded 32x32 f16 attention matrix per (b,head,branch) slot.
// ---------------------------------------------------------------------------
__global__ void attn_finalize(const float* __restrict__ partials,
                              const float* __restrict__ temp1, const float* __restrict__ temp2,
                              _Float16* __restrict__ attnOut) {
    const int slot = blockIdx.x;                  // b*16 + (head + 8*br)
    const int y = slot & 15, head = y & 7, br = y >> 3;
    const int t = threadIdx.x;

    __shared__ float red[1024];
    const float* p = partials + (size_t)slot * 16 * 1024;
    for (int i = t; i < 1024; i += 256) {
        float s = 0;
#pragma unroll
        for (int c = 0; c < 16; ++c) s += p[c * 1024 + i];
        red[i] = s;
    }
    __syncthreads();

    const float tmp = (br ? temp2 : temp1)[head];
    _Float16* dst = attnOut + (size_t)slot * 1024;
    if (t < HCH) {
        float v[HCH]; float m = -1e30f;
        for (int d = 0; d < HCH; ++d) { v[d] = red[t * 32 + d] * tmp; m = fmaxf(m, v[d]); }
        float sum = 0;
        for (int d = 0; d < HCH; ++d) { v[d] = __expf(v[d] - m); sum += v[d]; }
        float inv = 1.0f / sum;
        for (int d = 0; d < HCH; ++d) dst[t * 32 + d] = (_Float16)(v[d] * inv);
        for (int d = HCH; d < 32; ++d) dst[t * 32 + d] = (_Float16)0.0f;
    } else if (t < 32) {
        for (int d = 0; d < 32; ++d) dst[t * 32 + d] = (_Float16)0.0f;
    }
}

// ---------------------------------------------------------------------------
// out = attn @ v.  M=K=24 -> WMMA utilization would be ~9%; this stage is
// bound by streaming v (96 MB), so VALU: thread = pixel, 24 coalesced v
// loads, 576 FMAs against LDS-broadcast attn.
// ---------------------------------------------------------------------------
__global__ void attn_apply(const float* __restrict__ kv1, const float* __restrict__ kv2,
                           const _Float16* __restrict__ attn, float* __restrict__ outcat) {
    const int t = threadIdx.x;
    const int n = blockIdx.x * 256 + t;
    const int head = blockIdx.y & 7, br = blockIdx.y >> 3, b = blockIdx.z;

    __shared__ float A[HCH][HCH + 1];
    const _Float16* asrc = attn + ((size_t)((b * 2 + br) * HEADS + head)) * 1024;
    for (int i = t; i < HCH * HCH; i += 256) {
        int r = i / HCH, d = i % HCH;
        A[r][d] = (float)asrc[r * 32 + d];
    }
    __syncthreads();

    const float* vptr = (br ? kv2 : kv1) + ((size_t)(b * D2 + C + head * HCH)) * HW + n;
    float vr[HCH];
#pragma unroll
    for (int d = 0; d < HCH; ++d) vr[d] = vptr[(size_t)d * HW];

    float* op = outcat + ((size_t)(b * D2 + br * C + head * HCH)) * HW + n;
#pragma unroll
    for (int c = 0; c < HCH; ++c) {
        float s = 0;
#pragma unroll
        for (int d = 0; d < HCH; ++d) s += A[c][d] * vr[d];
        op[(size_t)c * HW] = s;
    }
}

// ---------------------------------------------------------------------------
// Host-side orchestration (all launches on `stream`; graph-capture safe).
// ---------------------------------------------------------------------------
extern "C" void kernel_launch(void* const* d_in, const int* in_sizes, int n_in,
                              void* d_out, int out_size, void* d_ws, size_t ws_size,
                              hipStream_t stream) {
    const float* x     = (const float*)d_in[0];
    const float* akv1  = (const float*)d_in[1];
    const float* akv2  = (const float*)d_in[2];
    const float* w_q   = (const float*)d_in[3];
    const float* w_kv1 = (const float*)d_in[4];
    const float* w_kv2 = (const float*)d_in[5];
    const float* w_qdw = (const float*)d_in[6];
    const float* w_k1d = (const float*)d_in[7];
    const float* w_k2d = (const float*)d_in[8];
    const float* w_prj = (const float*)d_in[9];
    const float* temp1 = (const float*)d_in[10];
    const float* temp2 = (const float*)d_in[11];

    // workspace layout (floats): 4 big [4,384,HW] buffers + norms/attn/partials
    float* ws = (float*)d_ws;
    const size_t SZ = (size_t)4 * D2 * HW;           // 25,165,824 floats (96 MB)
    float*     bufPre = ws;                          // scratch; reused as out_cat
    float*     qb     = ws + SZ;
    float*     kv1b   = ws + 2 * SZ;
    float*     kv2b   = ws + 3 * SZ;
    float*     normsB = ws + 4 * SZ;                 // 3072 floats
    _Float16*  attnB  = (_Float16*)(normsB + 4096);  // 64 * 1024 halves
    float*     partB  = (float*)(attnB + 64 * 1024); // 64*16*1024 floats (4 MB)

    const dim3 blk(256);
    const size_t shProj = (size_t)(32 * C  + 2 * 256 * BP) * sizeof(_Float16); // 52 KB
    const size_t shOut  = (size_t)(32 * D2 + 2 * 256 * BP) * sizeof(_Float16); // 64 KB
    const dim3 gGemmA(HW / 256, D2 / 32, 4);   // M=384, K=192
    const dim3 gGemmP(HW / 256, C  / 32, 4);   // M=192, K=384
    const dim3 gDw((unsigned)((size_t)4 * D2 * HW / 256));

    // 1x1 projections + dwconvs (bufPre reused between branches)
    gemm1x1  <<<gGemmA, blk, shProj, stream>>>(x,    w_q,   bufPre, D2, C);
    dwconv3x3<<<gDw,    blk, 0,      stream>>>(bufPre, w_qdw, qb,   2);
    gemm1x1  <<<gGemmA, blk, shProj, stream>>>(akv1, w_kv1, bufPre, D2, C);
    dwconv3x3<<<gDw,    blk, 0,      stream>>>(bufPre, w_k1d, kv1b, 1);
    gemm1x1  <<<gGemmA, blk, shProj, stream>>>(akv2, w_kv2, bufPre, D2, C);
    dwconv3x3<<<gDw,    blk, 0,      stream>>>(bufPre, w_k2d, kv2b, 1);

    // l2 norms, attention scores (partial + finalize), attn @ v -> bufPre
    rownorms     <<<dim3(3072),        blk, 0, stream>>>(qb, kv1b, kv2b, normsB);
    attn_partial <<<dim3(16, 16, 4),   blk, 0, stream>>>(qb, kv1b, kv2b, normsB, partB);
    attn_finalize<<<dim3(64),          blk, 0, stream>>>(partB, temp1, temp2, attnB);
    attn_apply   <<<dim3(HW / 256, 16, 4), blk, 0, stream>>>(kv1b, kv2b, attnB, bufPre);

    // final 1x1 projection -> d_out
    gemm1x1<<<gGemmP, blk, shOut, stream>>>(bufPre, w_prj, (float*)d_out, C, D2);
}
